// SinkhornLoss_65644280152782
// MI455X (gfx1250) — matrix operand order repaired and checked by reading, not compiled, not run on hardware
//
#include <hip/hip_runtime.h>
#include <hip/hip_bf16.h>

// SinkhornLoss for MI455X (gfx1250, wave32).
// ws layout (needs ~137 MB):
//   [0)        Xn  : 4096*512 f16  (4 MB)   row-normalized x
//   [4MB)      Yn  : 4096*512 f16  (4 MB)   row-normalized y
//   [8MB)      K   : 4096*4096 f32 (64 MB)  K = exp((cos_sim - 1)/EPS)
//   [72MB)     KT  : 4096*4096 f32 (64 MB)  K transposed (for coalesced K^T@a)
//   [136MB)    a   : 4096 f32
//   [136MB+16K)b   : 4096 f32

typedef __attribute__((ext_vector_type(16))) _Float16 v16h;
typedef __attribute__((ext_vector_type(8)))  _Float16 v8h;
typedef __attribute__((ext_vector_type(4)))  _Float16 v4h;
typedef __attribute__((ext_vector_type(8)))  float    v8f;

#define COLS  512
#define EPS_F 500.0f

__device__ __forceinline__ float wave_reduce_add(float v) {
#pragma unroll
    for (int off = 16; off > 0; off >>= 1) v += __shfl_xor(v, off, 32);
    return v;
}

// One wave (32 lanes) per row of 512 f32: compute 1/max(||row||,1e-8), emit f16.
__global__ __launch_bounds__(256) void normalize_f16_kernel(const float* __restrict__ X,
                                                            _Float16* __restrict__ Xn) {
    const int row  = blockIdx.x * 8 + (threadIdx.x >> 5);
    const int lane = threadIdx.x & 31;
    const float* xr = X + (size_t)row * COLS + lane * 16;
    float4 v[4];
    float ssq = 0.f;
#pragma unroll
    for (int t = 0; t < 4; ++t) {
        v[t] = *(const float4*)(xr + t * 4);
        ssq += v[t].x * v[t].x + v[t].y * v[t].y + v[t].z * v[t].z + v[t].w * v[t].w;
    }
    ssq = wave_reduce_add(ssq);
    const float inv = 1.0f / fmaxf(sqrtf(ssq), 1e-8f);
    _Float16* orow = Xn + (size_t)row * COLS + lane * 16;
#pragma unroll
    for (int t = 0; t < 4; ++t) {
        v4h h;
        h[0] = (_Float16)(v[t].x * inv);
        h[1] = (_Float16)(v[t].y * inv);
        h[2] = (_Float16)(v[t].z * inv);
        h[3] = (_Float16)(v[t].w * inv);
        *(v4h*)(orow + t * 4) = h;
    }
}

// 8 waves/block, each wave one 16x16 tile (block covers 32 rows x 64 cols).
// S = Xn @ Yn^T (cosine similarity), K = exp((S-1)/EPS), store K and K^T.
__global__ __launch_bounds__(256) void gemm_expk_kernel(const _Float16* __restrict__ Xn,
                                                        const _Float16* __restrict__ Yn,
                                                        float* __restrict__ Km,
                                                        float* __restrict__ KTm,
                                                        int n) {
    const int wave = threadIdx.x >> 5;
    const int lane = threadIdx.x & 31;
    const int half = lane >> 4;     // 0: lanes 0-15, 1: lanes 16-31
    const int l16  = lane & 15;
    const int row0 = blockIdx.y * 32 + (wave >> 2) * 16;
    const int col0 = blockIdx.x * 64 + (wave & 3) * 16;

    // A-frag (16x32 f16): lane half 0 -> K in {0..7}+{16..23}, half 1 -> {8..15}+{24..31}
    const _Float16* arow = Xn + (size_t)(row0 + l16) * COLS + half * 8;
    // B-frag (32x16 f16): column n = row n of Yn; half 0 -> K in {0..7}+{16..23}, half 1 -> {8..15}+{24..31}
    const _Float16* brow = Yn + (size_t)(col0 + l16) * COLS + half * 8;

    v8f acc = {};
#pragma unroll 4
    for (int k0 = 0; k0 < COLS; k0 += 32) {
        union { v16h v; v8h h[2]; } A, B;
        A.h[0] = *(const v8h*)(arow + k0);
        A.h[1] = *(const v8h*)(arow + k0 + 16);
        B.h[0] = *(const v8h*)(brow + k0);
        B.h[1] = *(const v8h*)(brow + k0 + 16);
        acc = __builtin_amdgcn_wmma_f32_16x16x32_f16(false, A.v, false, B.v,
                                                     (short)0, acc, false, false);
    }
    // D layout: VGPR r -> row (r + 8*half), col = l16
#pragma unroll
    for (int r = 0; r < 8; ++r) {
        const int row = row0 + half * 8 + r;
        const int col = col0 + l16;
        const float kv = expf((acc[r] - 1.0f) * (1.0f / EPS_F));
        Km [(size_t)row * n + col] = kv;
        KTm[(size_t)col * n + row] = kv;
    }
}

__global__ __launch_bounds__(256) void init_kernel(float* __restrict__ a,
                                                   float* __restrict__ b,
                                                   float* __restrict__ out,
                                                   int n, float invn) {
    const int i = blockIdx.x * blockDim.x + threadIdx.x;
    if (i < n) { a[i] = invn; b[i] = invn; }
    if (i == 0) out[0] = 0.0f;
}

// out[row] = invn / dot(M[row,:], v).  One wave per row, float4 streaming.
// M (64 MB) stays resident in the 192 MB L2 across all 1000 calls.
__global__ __launch_bounds__(256) void gemv_recip_kernel(const float* __restrict__ M,
                                                         const float* __restrict__ v,
                                                         float* __restrict__ out,
                                                         int n, float invn) {
    const int row  = blockIdx.x * 8 + (threadIdx.x >> 5);
    const int lane = threadIdx.x & 31;
    const float* mrow = M + (size_t)row * n;
    float acc = 0.f;
    for (int j = lane * 4; j < n; j += 128) {
        __builtin_prefetch(mrow + j + 1024, 0, 0);   // global_prefetch_b8
        const float4 m4 = *(const float4*)(mrow + j);
        const float4 v4 = *(const float4*)(v + j);
        acc = fmaf(m4.x, v4.x, acc);
        acc = fmaf(m4.y, v4.y, acc);
        acc = fmaf(m4.z, v4.z, acc);
        acc = fmaf(m4.w, v4.w, acc);
    }
    acc = wave_reduce_add(acc);
    if (lane == 0) out[row] = invn / acc;
}

// loss = sum_i a_i * sum_j K_ij * C_ij * b_j, with C = -EPS * ln(K).
__global__ __launch_bounds__(256) void loss_kernel(const float* __restrict__ Km,
                                                   const float* __restrict__ a,
                                                   const float* __restrict__ b,
                                                   float* __restrict__ out, int n) {
    const int row  = blockIdx.x * 8 + (threadIdx.x >> 5);
    const int lane = threadIdx.x & 31;
    const float* krow = Km + (size_t)row * n;
    float acc = 0.f;
    for (int j = lane * 4; j < n; j += 128) {
        const float4 k4 = *(const float4*)(krow + j);
        const float4 b4 = *(const float4*)(b + j);
        acc += k4.x * (-EPS_F * logf(k4.x)) * b4.x;
        acc += k4.y * (-EPS_F * logf(k4.y)) * b4.y;
        acc += k4.z * (-EPS_F * logf(k4.z)) * b4.z;
        acc += k4.w * (-EPS_F * logf(k4.w)) * b4.w;
    }
    acc = wave_reduce_add(acc);
    if (lane == 0) atomicAdd(out, a[row] * acc);
}

extern "C" void kernel_launch(void* const* d_in, const int* in_sizes, int n_in,
                              void* d_out, int out_size, void* d_ws, size_t ws_size,
                              hipStream_t stream) {
    const float* x = (const float*)d_in[0];
    const float* y = (const float*)d_in[1];
    float* out = (float*)d_out;

    const int n = in_sizes[0] / COLS;          // 4096
    const float invn = 1.0f / (float)n;

    char* ws = (char*)d_ws;
    const size_t half_bytes = (size_t)n * COLS * sizeof(_Float16);
    _Float16* Xn = (_Float16*)ws;
    _Float16* Yn = (_Float16*)(ws + half_bytes);
    float* Km = (float*)(ws + 2 * half_bytes);
    float* KT = Km + (size_t)n * n;
    float* a  = KT + (size_t)n * n;
    float* b  = a + n;

    // 1) row-normalize to f16
    normalize_f16_kernel<<<n / 8, 256, 0, stream>>>(x, Xn);
    normalize_f16_kernel<<<n / 8, 256, 0, stream>>>(y, Yn);

    // 2) WMMA GEMM + exp epilogue -> K, K^T
    dim3 g(n / 64, n / 32);
    gemm_expk_kernel<<<g, 256, 0, stream>>>(Xn, Yn, Km, KT, n);

    // 3) Sinkhorn iterations (a = 1/n / (K b); b = 1/n / (K^T a))
    init_kernel<<<(n + 255) / 256, 256, 0, stream>>>(a, b, out, n, invn);
    for (int it = 0; it < 500; ++it) {
        gemv_recip_kernel<<<n / 8, 256, 0, stream>>>(Km, b, a, n, invn);
        gemv_recip_kernel<<<n / 8, 256, 0, stream>>>(KT, a, b, n, invn);
    }

    // 4) loss = a^T (K .* C) b
    loss_kernel<<<n / 8, 256, 0, stream>>>(Km, a, b, out, n);
}
